// STMetaSelfAttentionLayer_27693949125282
// MI455X (gfx1250) — matrix operand order repaired and checked by reading, not compiled, not run on hardware
//
#include <hip/hip_runtime.h>
#include <hip/hip_bf16.h>

typedef __bf16 bf16_t;
typedef __attribute__((ext_vector_type(16))) __bf16 v16bf;
typedef __attribute__((ext_vector_type(8)))  float  v8f;
typedef __attribute__((ext_vector_type(4))) unsigned int u32x4;
typedef __attribute__((ext_vector_type(8))) int          i32x8;
typedef __attribute__((ext_vector_type(4))) int          i32x4;

#if defined(__has_builtin)
#  if __has_builtin(__builtin_amdgcn_tensor_load_to_lds) && __has_builtin(__builtin_amdgcn_s_wait_tensorcnt)
#    define HAVE_TDM 1
#  endif
#endif
#ifndef HAVE_TDM
#  define HAVE_TDM 0
#endif

#define B_   64
#define T_   12
#define N_   207
#define D_   64
#define STZ  127
#define LH_  128
#define FF_  128
#define H_   4
#define HD_  16
#define R_   (B_ * N_)       // 13248 rows (b,n)
#define K3D  (3 * D_)        // 192
#define WCOLS (D_ * K3D)     // 12288

// ---------------------------------------------------------------------------
// WMMA fragment helpers (CDNA5 wave32, v_wmma_f32_16x16x32_bf16)
// A-fragment (16x32 bf16, row-major, ld % 8 == 0): two 16B loads per lane.
// B-fragment from TRANSPOSED (N-major) storage: lane holds K kb..kb+7, kb+16..kb+23 (kb=0/8)
//   for its column N -> two 16B loads per lane (ds/global_load_b128).
// C/D: VGPR e: lanes 0-15 M=e, lanes 16-31 M=e+8; N=lane&15.
// ---------------------------------------------------------------------------
__device__ __forceinline__ v16bf load_a_frag(const bf16_t* p, int ld, int lane) {
  const int m  = lane & 15;
  const int kb = (lane < 16) ? 0 : 8;
  const bf16_t* row = p + m * ld + kb;
  union { v16bf v; uint4 q[2]; } u;
  u.q[0] = *(const uint4*)(row);        // K = kb+0 .. kb+7
  u.q[1] = *(const uint4*)(row + 16);   // K = kb+16 .. kb+23
  return u.v;
}

__device__ __forceinline__ v16bf load_b_frag_t(const bf16_t* pT, int ld, int lane) {
  const int n  = lane & 15;
  const int kb = (lane < 16) ? 0 : 8;
  const bf16_t* src = pT + n * ld + kb;
  union { v16bf v; uint4 q[2]; } u;
  u.q[0] = *(const uint4*)(src);        // K = kb+0 .. kb+7
  u.q[1] = *(const uint4*)(src + 16);   // K = kb+16 .. kb+23
  return u.v;
}

__device__ __forceinline__ v8f wmma_bf16(v16bf a, v16bf b, v8f c) {
  return __builtin_amdgcn_wmma_f32_16x16x32_bf16(
      /*neg_a=*/false, a, /*neg_b=*/false, b,
      /*c_mod=*/(short)0, c, /*reuse_a=*/false, /*reuse_b=*/false);
}

// ---------------------------------------------------------------------------
// Kernel 0: convert shared weights to bf16; Wo/f1/f2 stored TRANSPOSED (N,K)
// ---------------------------------------------------------------------------
__global__ void convert_weights(const float* __restrict__ lw2W,
                                const float* __restrict__ Wo,
                                const float* __restrict__ f1W,
                                const float* __restrict__ f2W,
                                bf16_t* __restrict__ lw2B,
                                bf16_t* __restrict__ WoT,
                                bf16_t* __restrict__ f1T,
                                bf16_t* __restrict__ f2T) {
  const int gid    = blockIdx.x * blockDim.x + threadIdx.x;
  const int stride = gridDim.x * blockDim.x;
  for (int i = gid; i < LH_ * WCOLS; i += stride) lw2B[i] = (bf16_t)lw2W[i];
  if (gid < D_ * D_) {               // Wo (k,n) -> WoT (n,k), ld = D_
    int kk = gid / D_, nn = gid % D_;
    WoT[nn * D_ + kk] = (bf16_t)Wo[gid];
  }
  if (gid < D_ * FF_) {              // f1 (k=D, n=FF) -> f1T (n,k), ld = D_
    int kk = gid / FF_, nn = gid % FF_;
    f1T[nn * D_ + kk] = (bf16_t)f1W[gid];
  }
  if (gid < FF_ * D_) {              // f2 (k=FF, n=D) -> f2T (n,k), ld = FF_
    int kk = gid / D_, nn = gid % D_;
    f2T[nn * FF_ + kk] = (bf16_t)f2W[gid];
  }
}

// ---------------------------------------------------------------------------
// Kernel 1: hypernetwork front: h_w = relu(meta@lw1+b) (bf16),
//           h_b = relu(meta@lb1+b), bqkv = h_b@lb2 + lb2_b (f32)
// ---------------------------------------------------------------------------
__global__ void hyper_front(const float* __restrict__ meta,
                            const float* __restrict__ lw1W, const float* __restrict__ lw1b,
                            const float* __restrict__ lb1W, const float* __restrict__ lb1b,
                            const float* __restrict__ lb2W, const float* __restrict__ lb2b,
                            bf16_t* __restrict__ hwB, float* __restrict__ bqkv) {
  __shared__ float hb[LH_];
  __shared__ float mrow[STZ];
  const int tid = threadIdx.x;
  const int r   = blockIdx.x;

  for (int i = tid; i < STZ; i += 128) mrow[i] = meta[(size_t)r * STZ + i];
  __syncthreads();

  float aw = lw1b[tid];
  float ab = lb1b[tid];
  for (int k = 0; k < STZ; ++k) {
    float m = mrow[k];
    aw = fmaf(m, lw1W[k * LH_ + tid], aw);
    ab = fmaf(m, lb1W[k * LH_ + tid], ab);
  }
  hwB[(size_t)r * LH_ + tid] = (bf16_t)fmaxf(aw, 0.0f);
  hb[tid] = fmaxf(ab, 0.0f);
  __syncthreads();

  for (int c = tid; c < K3D; c += 128) {
    float s = lb2b[c];
    for (int k = 0; k < LH_; ++k) s = fmaf(hb[k], lb2W[k * K3D + c], s);
    bqkv[(size_t)r * K3D + c] = s;
  }
}

// ---------------------------------------------------------------------------
// Kernel 2: big WMMA GEMM  Wbig = hwB(13248x128) @ lw2B(128x12288) + lw2_b
// block tile 64(M) x 128(N), K=128 single LDS pass.
// A tile staged by the Tensor Data Mover (pad_enable reproduces the +8
// element row padding); B tile transposed on the LDS store side.
// 256 threads = 8 waves, each wave -> 32x32 (2x2 wmma accumulators).
// ---------------------------------------------------------------------------
__global__ void __launch_bounds__(256)
hyper_gemm(const bf16_t* __restrict__ hwB, const bf16_t* __restrict__ lw2B,
           const float* __restrict__ lw2b, bf16_t* __restrict__ Wbig) {
  __shared__ __align__(16) bf16_t At[64 * 136];    // 64 rows(M) x 128(K), pad +8
  __shared__ __align__(16) bf16_t BtT[128 * 136];  // 128 rows(N) x 128(K), pad +8
  const int tid  = threadIdx.x;
  const int lane = tid & 31;
  const int wave = tid >> 5;
  const int n0   = blockIdx.x * 128;
  const int m0   = blockIdx.y * 64;

#if HAVE_TDM
  // ---- A tile via TDM: 2D tile 128(dim0) x 64(dim1) bf16, row stride 128,
  //      LDS pad: every 64 DWORDs (one 256B row) insert 4 DWORDs (16B = 8 elem)
  if (wave == 0) {
    unsigned long long ga = (unsigned long long)(size_t)(hwB + (size_t)m0 * LH_);
    unsigned int lds_off  = (unsigned int)(size_t)(void*)At;  // low 32 bits = LDS byte addr
    u32x4 g0;
    g0[0] = 1u;                                   // count=1 (valid), user mode
    g0[1] = lds_off;                              // lds_addr
    g0[2] = (unsigned int)ga;                     // global_addr[31:0]
    g0[3] = (unsigned int)(ga >> 32) | (2u << 30);// global_addr[56:32] | type=2
    i32x8 g1;
    g1[0] = (int)((1u << 16) |                    // data_size = 2 bytes
                  (1u << 20) |                    // pad_enable
                  (5u << 22) |                    // pad_interval: 64 DWORDs
                  (3u << 25));                    // pad_amount: 4 DWORDs
    g1[1] = (int)(128u << 16);                    // tensor_dim0 = 128 elems
    g1[2] = (int)(64u << 16);                     // tensor_dim1 = 64 rows
    g1[3] = (int)(128u << 16);                    // tile_dim0 = 128
    g1[4] = 64;                                   // tile_dim1 = 64
    g1[5] = 128;                                  // tensor_dim0_stride = 128 elems
    g1[6] = 0;
    g1[7] = 0;
    i32x4 g2 = {0, 0, 0, 0};
    i32x4 g3 = {0, 0, 0, 0};
#if defined(__clang_major__) && __clang_major__ >= 23
    i32x8 g4 = {0, 0, 0, 0, 0, 0, 0, 0};
    __builtin_amdgcn_tensor_load_to_lds(g0, g1, g2, g3, g4, 0);
#else
    __builtin_amdgcn_tensor_load_to_lds(g0, g1, g2, g3, 0);
#endif
  }
#else
  // ---- A tile: 1024 uint4 (row-major copy)
#pragma unroll
  for (int i = 0; i < 4; ++i) {
    int v = tid + i * 256;
    int row = v >> 4, cv = v & 15;
    *(uint4*)(At + row * 136 + cv * 8) =
        *(const uint4*)(hwB + (size_t)(m0 + row) * LH_ + cv * 8);
  }
#endif

  // ---- B tile: 2048 uint4 read, transpose on the LDS store side
#pragma unroll
  for (int i = 0; i < 8; ++i) {
    int v = tid + i * 256;
    int krow = v >> 4, cv = v & 15;
    uint4 d4 = *(const uint4*)(lw2B + (size_t)krow * WCOLS + n0 + cv * 8);
    const bf16_t* e = (const bf16_t*)&d4;
#pragma unroll
    for (int j = 0; j < 8; ++j)
      BtT[(cv * 8 + j) * 136 + krow] = e[j];
  }

#if HAVE_TDM
  if (wave == 0) __builtin_amdgcn_s_wait_tensorcnt(0);
#endif
  __syncthreads();

  const int wm = wave >> 2;   // 0..1  (M group of 32)
  const int wn = wave & 3;    // 0..3  (N group of 32)
  v8f acc[2][2] = {};
#pragma unroll
  for (int ks = 0; ks < 4; ++ks) {
    v16bf a0 = load_a_frag(At + (wm * 32 +  0) * 136 + ks * 32, 136, lane);
    v16bf a1 = load_a_frag(At + (wm * 32 + 16) * 136 + ks * 32, 136, lane);
    v16bf b0 = load_b_frag_t(BtT + (wn * 32 +  0) * 136 + ks * 32, 136, lane);
    v16bf b1 = load_b_frag_t(BtT + (wn * 32 + 16) * 136 + ks * 32, 136, lane);
    acc[0][0] = wmma_bf16(a0, b0, acc[0][0]);
    acc[0][1] = wmma_bf16(a0, b1, acc[0][1]);
    acc[1][0] = wmma_bf16(a1, b0, acc[1][0]);
    acc[1][1] = wmma_bf16(a1, b1, acc[1][1]);
  }

  const int coln = lane & 15;
  const int mofs = (lane < 16) ? 0 : 8;
#pragma unroll
  for (int i = 0; i < 2; ++i) {
#pragma unroll
    for (int j = 0; j < 2; ++j) {
      int colg   = n0 + wn * 32 + j * 16 + coln;
      float bias = lw2b[colg];
#pragma unroll
      for (int e = 0; e < 8; ++e) {
        int m = m0 + wm * 32 + i * 16 + e + mofs;
        Wbig[(size_t)m * WCOLS + colg] = (bf16_t)(acc[i][j][e] + bias);
      }
    }
  }
}

// ---------------------------------------------------------------------------
// Kernel 3: fused per-(b,n) epilogue. One block (128 thr = 4 waves) per row.
// qkv (WMMA) -> per-head attention (VALU) -> Wo proj (WMMA) -> LN1 ->
// FFN (2x WMMA) -> LN2 -> transposed store. Everything stays in LDS.
// Per-token W tile staged TRANSPOSED (k-major) so B-frags are ds_load_b128.
// ---------------------------------------------------------------------------
__global__ void __launch_bounds__(128)
fused_epilogue(const float* __restrict__ x,
               const bf16_t* __restrict__ Wbig,
               const float* __restrict__ bqkv,
               const bf16_t* __restrict__ WoT, const float* __restrict__ bo,
               const bf16_t* __restrict__ f1T, const float* __restrict__ f1b,
               const bf16_t* __restrict__ f2T, const float* __restrict__ f2b,
               const float* __restrict__ g1, const float* __restrict__ be1,
               const float* __restrict__ g2, const float* __restrict__ be2,
               float* __restrict__ out) {
  __shared__ __align__(16) unsigned char smem[62720];
  bf16_t* WtT   = (bf16_t*)(smem + 0);      // 192(k) x 72(d)  transposed W tile
  bf16_t* xb    = (bf16_t*)(smem + 27648);  // 16 x 72 bf16 (rows 12..15 zero)
  float*  xf    = (float*) (smem + 29952);  // 12 x 64 f32 residual
  float*  qkv   = (float*) (smem + 33024);  // 12 x 192 f32
  float*  score = (float*) (smem + 42240);  // 4 heads x 12 x 12
  bf16_t* aoB   = (bf16_t*)(smem + 44544);  // 16 x 72 attn out bf16
  float*  y1    = (float*) (smem + 46848);  // 12 x 64 pre-LN1
  float*  yln   = (float*) (smem + 49920);  // 12 x 64 post-LN1 (residual2)
  bf16_t* yB    = (bf16_t*)(smem + 52992);  // 16 x 72 post-LN1 bf16
  bf16_t* h1B   = (bf16_t*)(smem + 55296);  // 16 x 136 FFN hidden bf16
  float*  z     = (float*) (smem + 59648);  // 12 x 64 pre-LN2

  const int tid  = threadIdx.x;
  const int lane = tid & 31;
  const int w    = tid >> 5;     // wave id 0..3 (also attention head id)
  const int r    = blockIdx.x;
  const int b    = r / N_;
  const int n    = r % N_;

  if (r + 1 < R_)  // hint next row's W tile toward the caches
    __builtin_prefetch(Wbig + (size_t)(r + 1) * WCOLS, 0, 1);

  // zero-pad rows 12..15 of bf16 M-tiles
  for (int i = tid; i < 4 * 72; i += 128) {
    int rr = 12 + i / 72, c = i % 72;
    xb[rr * 72 + c]  = (bf16_t)0.0f;
    aoB[rr * 72 + c] = (bf16_t)0.0f;
    yB[rr * 72 + c]  = (bf16_t)0.0f;
  }
  // load x[b, :, n, :] -> xf (f32) and xb (bf16)
  for (int i = tid; i < T_ * D_; i += 128) {
    int t = i >> 6, d = i & 63;
    float v = x[(((size_t)b * T_ + t) * N_ + n) * D_ + d];
    xf[i] = v;
    xb[t * 72 + d] = (bf16_t)v;
  }
  // load per-token W (64x192, d-major) and transpose into WtT (k-major)
  {
    const uint4* src = (const uint4*)(Wbig + (size_t)r * WCOLS);
    for (int i = tid; i < 1536; i += 128) {
      int d = i / 24, c8 = i % 24;
      uint4 d4 = src[i];
      const bf16_t* e = (const bf16_t*)&d4;
#pragma unroll
      for (int j = 0; j < 8; ++j)
        WtT[(c8 * 8 + j) * 72 + d] = e[j];
    }
  }
  __syncthreads();

  const int coln = lane & 15;
  const int mofs = (lane < 16) ? 0 : 8;

  // ---- qkv = xb @ W + bqkv[r]   (M=16, N=192 -> 3 tiles/wave, K=64)
  {
    v8f acc[3] = {};
#pragma unroll
    for (int ks = 0; ks < 2; ++ks) {
      v16bf a = load_a_frag(xb + ks * 32, 72, lane);
#pragma unroll
      for (int j = 0; j < 3; ++j) {
        v16bf bf = load_b_frag_t(WtT + ((w * 3 + j) * 16) * 72 + ks * 32, 72, lane);
        acc[j] = wmma_bf16(a, bf, acc[j]);
      }
    }
    const float* bq = bqkv + (size_t)r * K3D;
#pragma unroll
    for (int j = 0; j < 3; ++j) {
      int col = (w * 3 + j) * 16 + coln;
      float bb = bq[col];
#pragma unroll
      for (int e = 0; e < 8; ++e) {
        int m = e + mofs;
        if (m < T_) qkv[m * K3D + col] = acc[j][e] + bb;
      }
    }
  }
  __syncthreads();

  // ---- attention, head w per wave: scores 12x12, softmax, @v
  for (int idx = lane; idx < T_ * T_; idx += 32) {
    int t = idx / T_, u = idx % T_;
    float s = 0.0f;
#pragma unroll
    for (int d2 = 0; d2 < HD_; ++d2)
      s += qkv[t * K3D + w * HD_ + d2] * qkv[u * K3D + D_ + w * HD_ + d2];
    score[w * 144 + idx] = s * 0.25f;   // 1/sqrt(16)
  }
  __syncthreads();
  if (lane < T_) {
    float* sr = score + w * 144 + lane * T_;
    float mx = sr[0];
    for (int u = 1; u < T_; ++u) mx = fmaxf(mx, sr[u]);
    float sum = 0.0f;
    for (int u = 0; u < T_; ++u) { float e = __expf(sr[u] - mx); sr[u] = e; sum += e; }
    float inv = 1.0f / sum;
    for (int u = 0; u < T_; ++u) sr[u] *= inv;
  }
  __syncthreads();
  for (int idx = lane; idx < T_ * HD_; idx += 32) {
    int t = idx / HD_, c = idx % HD_;
    float o = 0.0f;
    for (int u = 0; u < T_; ++u)
      o += score[w * 144 + t * T_ + u] * qkv[u * K3D + 2 * D_ + w * HD_ + c];
    aoB[t * 72 + w * HD_ + c] = (bf16_t)o;
  }
  __syncthreads();

  // ---- proj = aoB @ Wo + bo ; y1 = residual(xf) + proj   (N=64 -> 1 tile/wave)
  {
    v8f acc = {};
#pragma unroll
    for (int ks = 0; ks < 2; ++ks) {
      v16bf a  = load_a_frag(aoB + ks * 32, 72, lane);
      v16bf bf = load_b_frag_t(WoT + (w * 16) * D_ + ks * 32, D_, lane);
      acc = wmma_bf16(a, bf, acc);
    }
    int col = w * 16 + coln;
    float bb = bo[col];
#pragma unroll
    for (int e = 0; e < 8; ++e) {
      int m = e + mofs;
      if (m < T_) y1[m * D_ + col] = xf[m * D_ + col] + acc[e] + bb;
    }
  }
  __syncthreads();

  // ---- LN1
  if (tid < T_) {
    const float* row = y1 + tid * D_;
    float mu = 0.0f;
    for (int d = 0; d < D_; ++d) mu += row[d];
    mu *= (1.0f / D_);
    float var = 0.0f;
    for (int d = 0; d < D_; ++d) { float df = row[d] - mu; var += df * df; }
    var *= (1.0f / D_);
    float inv = rsqrtf(var + 1e-5f);
    for (int d = 0; d < D_; ++d) {
      float v = (row[d] - mu) * inv * g1[d] + be1[d];
      yln[tid * D_ + d] = v;
      yB[tid * 72 + d]  = (bf16_t)v;
    }
  }
  __syncthreads();

  // ---- FFN1: h1 = relu(yB @ f1 + f1b)   (N=128 -> 2 tiles/wave, K=64)
#pragma unroll
  for (int jj = 0; jj < 2; ++jj) {
    v8f acc = {};
    int nt = (w * 2 + jj) * 16;
#pragma unroll
    for (int ks = 0; ks < 2; ++ks) {
      v16bf a  = load_a_frag(yB + ks * 32, 72, lane);
      v16bf bf = load_b_frag_t(f1T + nt * D_ + ks * 32, D_, lane);
      acc = wmma_bf16(a, bf, acc);
    }
    int col = nt + coln;
    float bb1 = f1b[col];
#pragma unroll
    for (int e = 0; e < 8; ++e) {
      int m = e + mofs;
      h1B[m * 136 + col] = (bf16_t)fmaxf(acc[e] + bb1, 0.0f);
    }
  }
  __syncthreads();

  // ---- FFN2: z = yln + h1B @ f2 + f2b   (N=64 -> 1 tile/wave, K=128)
  {
    v8f acc = {};
#pragma unroll
    for (int ks = 0; ks < 4; ++ks) {
      v16bf a  = load_a_frag(h1B + ks * 32, 136, lane);
      v16bf bf = load_b_frag_t(f2T + (w * 16) * FF_ + ks * 32, FF_, lane);
      acc = wmma_bf16(a, bf, acc);
    }
    int col = w * 16 + coln;
    float bb2 = f2b[col];
#pragma unroll
    for (int e = 0; e < 8; ++e) {
      int m = e + mofs;
      if (m < T_) z[m * D_ + col] = yln[m * D_ + col] + acc[e] + bb2;
    }
  }
  __syncthreads();

  // ---- LN2 + transposed store to out[b, t, n, :]
  if (tid < T_) {
    const float* row = z + tid * D_;
    float mu = 0.0f;
    for (int d = 0; d < D_; ++d) mu += row[d];
    mu *= (1.0f / D_);
    float var = 0.0f;
    for (int d = 0; d < D_; ++d) { float df = row[d] - mu; var += df * df; }
    var *= (1.0f / D_);
    float inv = rsqrtf(var + 1e-5f);
    float* orow = out + (((size_t)b * T_ + tid) * N_ + n) * D_;
    for (int d = 0; d < D_; ++d)
      orow[d] = (row[d] - mu) * inv * g2[d] + be2[d];
  }
}

// ---------------------------------------------------------------------------
extern "C" void kernel_launch(void* const* d_in, const int* in_sizes, int n_in,
                              void* d_out, int out_size, void* d_ws, size_t ws_size,
                              hipStream_t stream) {
  (void)in_sizes; (void)n_in; (void)out_size; (void)ws_size;
  const float* x    = (const float*)d_in[0];
  const float* meta = (const float*)d_in[1];
  /* d_in[2] = spatial (==0, compile-time path) */
  const float* lw1W = (const float*)d_in[3];
  const float* lw1b = (const float*)d_in[4];
  const float* lw2W = (const float*)d_in[5];
  const float* lw2b = (const float*)d_in[6];
  const float* lb1W = (const float*)d_in[7];
  const float* lb1b = (const float*)d_in[8];
  const float* lb2W = (const float*)d_in[9];
  const float* lb2b = (const float*)d_in[10];
  const float* Wo   = (const float*)d_in[11];
  const float* bo   = (const float*)d_in[12];
  const float* f1W  = (const float*)d_in[13];
  const float* f1b  = (const float*)d_in[14];
  const float* f2W  = (const float*)d_in[15];
  const float* f2b  = (const float*)d_in[16];
  const float* g1   = (const float*)d_in[17];
  const float* be1  = (const float*)d_in[18];
  const float* g2   = (const float*)d_in[19];
  const float* be2  = (const float*)d_in[20];
  float* out = (float*)d_out;

  char* ws = (char*)d_ws;
  size_t off = 0;
  auto take = [&](size_t bytes) -> char* {
    char* p = ws + off;
    off = (off + bytes + 255) & ~(size_t)255;
    return p;
  };
  bf16_t* hwB  = (bf16_t*)take((size_t)R_ * LH_ * 2);
  float*  bq   = (float*) take((size_t)R_ * K3D * 4);
  bf16_t* lw2B = (bf16_t*)take((size_t)LH_ * WCOLS * 2);
  bf16_t* WoT  = (bf16_t*)take((size_t)D_ * D_ * 2);
  bf16_t* f1T  = (bf16_t*)take((size_t)D_ * FF_ * 2);
  bf16_t* f2T  = (bf16_t*)take((size_t)FF_ * D_ * 2);
  bf16_t* Wbig = (bf16_t*)take((size_t)R_ * WCOLS * 2);

  convert_weights<<<6144, 256, 0, stream>>>(lw2W, Wo, f1W, f2W, lw2B, WoT, f1T, f2T);
  hyper_front<<<R_, 128, 0, stream>>>(meta, lw1W, lw1b, lb1W, lb1b, lb2W, lb2b, hwB, bq);
  hyper_gemm<<<dim3(WCOLS / 128, R_ / 64), 256, 0, stream>>>(hwB, lw2B, lw2b, Wbig);
  fused_epilogue<<<R_, 128, 0, stream>>>(x, Wbig, bq, WoT, bo, f1T, f1b, f2T, f2b,
                                         g1, be1, g2, be2, out);
}
